// VQ_14499809591797
// MI455X (gfx1250) — matrix operand order repaired and checked by reading, not compiled, not run on hardware
//
#include <hip/hip_runtime.h>

typedef __attribute__((ext_vector_type(16))) _Float16 v16h;
typedef __attribute__((ext_vector_type(8)))  float    v8f;

#define CDIM        64
#define LWIDTH      4096
#define COLS_PER_WG 128
// loss = 1.25 * mean over (c=64, t=131072) = 1.25/8388608 per squared-diff
#define LOSS_SCALE  (1.25f / 8388608.0f)

// ------------------------------------------------------------------
// Prep 1: codes_emb (1024x64 f32) -> f16 WMMA A-fragments, scaled by -2.
// Fragment layout exactly matches ISA "16-bit A-Matrix 16x32" (wave32):
//   lane L holds code row M = L%16, half = L/16
//   vgpr j<4 : K = half*8 + 2j{,+1};  j>=4 : K = 16 + half*8 + 2(j-4){,+1}
// ws layout: frag index tid = cb*64 + kc*32 + lane, 16 f16 (32B) each.
// ------------------------------------------------------------------
__global__ void vq_prep_codes(const float* __restrict__ codes,
                              _Float16* __restrict__ codesA) {
  int tid  = blockIdx.x * blockDim.x + threadIdx.x;   // 0..4095
  int lane = tid & 31;
  int kc   = (tid >> 5) & 1;
  int cb   = tid >> 6;
  int half = lane >> 4;
  int code = cb * 16 + (lane & 15);
  _Float16* out = codesA + (size_t)tid * 16;
  #pragma unroll
  for (int j = 0; j < 8; ++j) {
    int K0  = (j < 4) ? (half * 8 + 2 * j) : (16 + half * 8 + 2 * (j - 4));
    int ch0 = kc * 32 + K0;
    out[2 * j]     = (_Float16)(-2.0f * codes[code * CDIM + ch0]);
    out[2 * j + 1] = (_Float16)(-2.0f * codes[code * CDIM + ch0 + 1]);
  }
}

// Prep 2: y2[d] = sum_c codes[d][c]^2  (1024 floats)
__global__ void vq_prep_y2(const float* __restrict__ codes,
                           float* __restrict__ y2) {
  int d = blockIdx.x * blockDim.x + threadIdx.x;
  float s = 0.f;
  #pragma unroll 8
  for (int c = 0; c < CDIM; ++c) { float v = codes[d * CDIM + c]; s += v * v; }
  y2[d] = s;
}

// ------------------------------------------------------------------
// Main: one workgroup = 128 columns (one b, one l-tile), 8 waves x 16 cols.
// Per wave: build B fragments (32ch x 16col f16) once from x, then sweep
// 64 code blocks: acc = y2 + (-2*codes)*x via v_wmma_f32_16x16x32_f16,
// running per-lane argmin; shfl-combine halves; LDS indices; phase-2 does
// coalesced gather/store of qs + loss accumulation.
// ------------------------------------------------------------------
__global__ __launch_bounds__(256) void vq_main(
    const float* __restrict__ x, const float* __restrict__ codes,
    const _Float16* __restrict__ codesA, const float* __restrict__ y2,
    float* __restrict__ loss, float* __restrict__ qs) {
  __shared__ int s_idx[COLS_PER_WG];

  const int wg   = blockIdx.x;             // 0..1023
  const int b    = wg >> 5;                // 32 l-tiles per batch
  const int l0   = (wg & 31) * COLS_PER_WG;
  const int tid  = threadIdx.x;
  const int w    = tid >> 5;
  const int lane = tid & 31;
  const int n    = lane & 15;
  const int half = lane >> 4;
  const int half8 = half << 3;
  const int col  = l0 + (w << 4) + n;

  // ---- build B fragments: same K order as A: vgpr j<4 : K = half*8+2j{,+1}; j>=4 : K = 16+half*8+2(j-4){,+1}
  const float* xb = x + ((size_t)b * CDIM) * LWIDTH + col;
  v16h bf0, bf1;
  #pragma unroll
  for (int j = 0; j < 8; ++j) {
    int c0 = (j < 4) ? (half * 8 + 2 * j) : (16 + half * 8 + 2 * (j - 4));
    bf0[2 * j]     = (_Float16)xb[(size_t)(c0)      * LWIDTH];
    bf0[2 * j + 1] = (_Float16)xb[(size_t)(c0 + 1)  * LWIDTH];
    bf1[2 * j]     = (_Float16)xb[(size_t)(c0 + 32) * LWIDTH];
    bf1[2 * j + 1] = (_Float16)xb[(size_t)(c0 + 33) * LWIDTH];
  }

  // ---- sweep 1024 codes in 64 blocks of 16
  float best   = __builtin_inff();
  int bestIdx  = 0;
  const v16h* af = (const v16h*)codesA + lane;   // +cb*64 (kc=0), +32 (kc=1)

  for (int cb = 0; cb < 64; ++cb) {
    v16h a0 = af[cb * 64];
    v16h a1 = af[cb * 64 + 32];
    if (cb + 2 < 64)
      __builtin_prefetch((const void*)(af + (cb + 2) * 64), 0, 3);

    // C init = ||y||^2 for this lane's 8 codes -> D = y2 - 2*x.y (the score)
    v8f acc = *(const v8f*)(y2 + cb * 16 + half8);
    acc = __builtin_amdgcn_wmma_f32_16x16x32_f16(false, a0, false, bf0,
                                                 (short)0, acc, false, false);
    acc = __builtin_amdgcn_wmma_f32_16x16x32_f16(false, a1, false, bf1,
                                                 (short)0, acc, false, false);

    int base = cb * 16 + half8;
    #pragma unroll
    for (int j = 0; j < 8; ++j) {
      float s = acc[j];
      if (s < best) { best = s; bestIdx = base + j; }
    }
  }

  // combine the two code halves (lanes L and L+16 hold the same column)
  float ob = __shfl_xor(best, 16, 32);
  int   oi = __shfl_xor(bestIdx, 16, 32);
  if (ob < best || (ob == best && oi < bestIdx)) { best = ob; bestIdx = oi; }
  if (half == 0) s_idx[(w << 4) + n] = bestIdx;
  __syncthreads();

  // ---- phase 2: coalesced qs gather/store + loss (re-reads x in f32)
  float lacc = 0.f;
  const size_t xbase = ((size_t)b * CDIM) * LWIDTH + l0;
  #pragma unroll 4
  for (int it = 0; it < 32; ++it) {
    int e   = it * 256 + tid;      // 0..8191 over (64 ch x 128 col)
    int ch  = e >> 7;
    int cc  = e & 127;
    int idx = s_idx[cc];
    float q = codes[idx * CDIM + ch];
    size_t off = xbase + (size_t)ch * LWIDTH + cc;
    float d = q - x[off];
    lacc += d * d;
    qs[off] = q;
  }
  #pragma unroll
  for (int o = 16; o > 0; o >>= 1) lacc += __shfl_xor(lacc, o, 32);
  if (lane == 0) atomicAdd(loss, lacc * LOSS_SCALE);
}

extern "C" void kernel_launch(void* const* d_in, const int* in_sizes, int n_in,
                              void* d_out, int out_size, void* d_ws, size_t ws_size,
                              hipStream_t stream) {
  const float* x     = (const float*)d_in[0];   // (32, 64, 4096) f32
  const float* codes = (const float*)d_in[1];   // (1024, 64) f32
  float* out = (float*)d_out;                   // [loss, qs...]

  _Float16* codesA = (_Float16*)d_ws;                           // 128 KB
  float*    y2     = (float*)((char*)d_ws + 4096 * 16 * 2);     // +4 KB

  hipMemsetAsync(d_out, 0, sizeof(float), stream);              // zero loss
  vq_prep_codes<<<16, 256, 0, stream>>>(codes, codesA);
  vq_prep_y2  <<<4,  256, 0, stream>>>(codes, y2);
  vq_main<<<1024, 256, 0, stream>>>(x, codes, codesA, y2, out, out + 1);
}
